// GNNLayer_78005196030507
// MI455X (gfx1250) — matrix operand order repaired and checked by reading, not compiled, not run on hardware
//
#include <hip/hip_runtime.h>
#include <hip/hip_bf16.h>

typedef __attribute__((ext_vector_type(16))) _Float16 v16h;
typedef __attribute__((ext_vector_type(8)))  _Float16 v8h;
typedef __attribute__((ext_vector_type(4)))  _Float16 v4h;
typedef __attribute__((ext_vector_type(8)))  float    v8f;
typedef __attribute__((ext_vector_type(4)))  float    v4f;

#define D 128
#define BN_EPS 1e-5f

// ---------------- zero init (deg | agg | gsum | gsumsq are contiguous) -------
__global__ void k_zero(float* __restrict__ p, int n) {
  int i = blockIdx.x * blockDim.x + threadIdx.x;
  if (i < n) p[i] = 0.0f;
}

// ---------------- W (f32) -> f16 ---------------------------------------------
__global__ void k_wconv(const float* __restrict__ W, _Float16* __restrict__ wf) {
  int i = blockIdx.x * blockDim.x + threadIdx.x;  // over D*D/4
  if (i >= (D * D) / 4) return;
  v4f w = reinterpret_cast<const v4f*>(W)[i];
  v4h o;
  o[0] = (_Float16)w[0]; o[1] = (_Float16)w[1];
  o[2] = (_Float16)w[2]; o[3] = (_Float16)w[3];
  reinterpret_cast<v4h*>(wf)[i] = o;
}

// ---------------- edge scatter: agg[row] += x[col]; deg[row] += 1 ------------
// one thread per (edge, 4-float chunk): 32 lanes cover one edge's 128 features
__global__ void k_scatter(const float* __restrict__ x, const int* __restrict__ row,
                          const int* __restrict__ col, float* __restrict__ agg,
                          float* __restrict__ deg, int nEdges) {
  int i = blockIdx.x * blockDim.x + threadIdx.x;
  int e = i >> 5;
  if (e >= nEdges) return;
  int c4 = (i & 31) * 4;
  int r = row[e], c = col[e];
  v4f xv = *reinterpret_cast<const v4f*>(x + (size_t)c * D + c4);
  float* dst = agg + (size_t)r * D + c4;
  atomicAdd(dst + 0, xv[0]);
  atomicAdd(dst + 1, xv[1]);
  atomicAdd(dst + 2, xv[2]);
  atomicAdd(dst + 3, xv[3]);
  if (c4 == 0) atomicAdd(deg + r, 1.0f);
}

// ---------------- h = x + agg / max(deg,1), stored f16 -----------------------
__global__ void k_h(const float* __restrict__ x, const float* __restrict__ agg,
                    const float* __restrict__ deg, _Float16* __restrict__ hf, int n4) {
  int i = blockIdx.x * blockDim.x + threadIdx.x;  // over N*D/4
  if (i >= n4) return;
  int node = i >> 5;
  float inv = 1.0f / fmaxf(deg[node], 1.0f);
  v4f xv = reinterpret_cast<const v4f*>(x)[i];
  v4f av = reinterpret_cast<const v4f*>(agg)[i];
  v4h h;
#pragma unroll
  for (int j = 0; j < 4; ++j) h[j] = (_Float16)(xv[j] + av[j] * inv);
  reinterpret_cast<v4h*>(hf)[i] = h;
}

// ---------------- WMMA GEMM: y = h @ W^T + b, fused BN-stat accumulation -----
// block = 8 waves; wave w computes 16x16 tile (rows blockIdx.x*16, cols w*16)
__global__ void __launch_bounds__(256) k_gemm(const _Float16* __restrict__ hf,
                                              const _Float16* __restrict__ wf,
                                              const float* __restrict__ bias,
                                              float* __restrict__ y,
                                              float* __restrict__ gsum,
                                              float* __restrict__ gsumsq) {
  const int wave = threadIdx.x >> 5;
  const int lane = threadIdx.x & 31;
  const int half = lane >> 4;
  const int lm   = lane & 15;
  const int mBase = blockIdx.x * 16;
  const int nBase = wave * 16;

  // C init with bias: lane holds column N = lm for all 8 row-slots
  float bv = bias[nBase + lm];
  v8f c;
#pragma unroll
  for (int r = 0; r < 8; ++r) c[r] = bv;

  const _Float16* aRow = hf + (size_t)(mBase + lm) * D;  // A: M = lm
  const _Float16* bRow = wf + (size_t)(nBase + lm) * D;  // B: N = lm (W row-major [O][K])

#pragma unroll
  for (int k0 = 0; k0 < D; k0 += 32) {
    // A lane layout (16-bit 16x32): K = k0 + 8*half + {0..7}, then +16
    v8h alo = *reinterpret_cast<const v8h*>(aRow + k0 + 8 * half);
    v8h ahi = *reinterpret_cast<const v8h*>(aRow + k0 + 8 * half + 16);
    // B lane layout (32x16): K = k0 + 8*half + {0..7}, then +16
    v8h blo = *reinterpret_cast<const v8h*>(bRow + k0 + 8 * half);
    v8h bhi = *reinterpret_cast<const v8h*>(bRow + k0 + 8 * half + 16);
    v16h a, bm;
#pragma unroll
    for (int j = 0; j < 8; ++j) {
      a[j] = alo[j]; a[j + 8] = ahi[j];
      bm[j] = blo[j]; bm[j + 8] = bhi[j];
    }
    c = __builtin_amdgcn_wmma_f32_16x16x32_f16(false, a, false, bm,
                                               (short)0, c, false, false);
  }

  // store y (f32) + per-column partial sums for BN
  float s = 0.0f, s2 = 0.0f;
#pragma unroll
  for (int r = 0; r < 8; ++r) {
    float v = c[r];
    y[(size_t)(mBase + r + 8 * half) * D + nBase + lm] = v;
    s += v;
    s2 += v * v;
  }
  float so  = __shfl_xor(s, 16, 32);
  float s2o = __shfl_xor(s2, 16, 32);
  if (half == 0) {
    atomicAdd(&gsum[nBase + lm],   s + so);
    atomicAdd(&gsumsq[nBase + lm], s2 + s2o);
  }
}

// ---------------- BN params: scale/shift per channel -------------------------
__global__ void k_bnparams(const float* __restrict__ gsum, const float* __restrict__ gsumsq,
                           const float* __restrict__ gamma, const float* __restrict__ beta,
                           float* __restrict__ scale, float* __restrict__ shift, float invN) {
  int o = threadIdx.x;
  if (o >= D) return;
  float mean = gsum[o] * invN;
  float var  = gsumsq[o] * invN - mean * mean;
  float sc   = gamma[o] * rsqrtf(var + BN_EPS);
  scale[o] = sc;
  shift[o] = beta[o] - mean * sc;
}

// ---------------- apply BN + ReLU in-place on d_out --------------------------
__global__ void k_bnapply(float* __restrict__ y, const float* __restrict__ scale,
                          const float* __restrict__ shift, int n4) {
  int i = blockIdx.x * blockDim.x + threadIdx.x;  // over N*D/4
  if (i >= n4) return;
  int c4 = (i & 31) * 4;
  v4f yv = reinterpret_cast<const v4f*>(y)[i];
  v4f ov;
#pragma unroll
  for (int j = 0; j < 4; ++j) {
    float v = yv[j] * scale[c4 + j] + shift[c4 + j];
    ov[j] = fmaxf(v, 0.0f);
  }
  reinterpret_cast<v4f*>(y)[i] = ov;
}

extern "C" void kernel_launch(void* const* d_in, const int* in_sizes, int n_in,
                              void* d_out, int out_size, void* d_ws, size_t ws_size,
                              hipStream_t stream) {
  const float* x     = (const float*)d_in[0];
  const int*   ei    = (const int*)d_in[1];
  const float* W     = (const float*)d_in[2];
  const float* b     = (const float*)d_in[3];
  const float* gamma = (const float*)d_in[4];
  const float* beta  = (const float*)d_in[5];

  const int nNodes = in_sizes[0] / D;   // 50000 (multiple of 16)
  const int nEdges = in_sizes[1] / 2;   // 600000
  const int* row = ei;
  const int* col = ei + nEdges;

  // workspace layout (all float counts multiples of 4 -> 16B alignment holds)
  char* ws = (char*)d_ws;
  float* deg    = (float*)ws;                            // [nNodes]
  float* agg    = deg + nNodes;                          // [nNodes*D]
  float* gsum   = agg + (size_t)nNodes * D;              // [D]
  float* gsumsq = gsum + D;                              // [D]
  float* scale  = gsumsq + D;                            // [D]
  float* shift  = scale + D;                             // [D]
  _Float16* hf  = (_Float16*)(shift + D);                // [nNodes*D] f16
  _Float16* wf  = hf + (size_t)nNodes * D;               // [D*D] f16

  float* y = (float*)d_out;

  const int zeroN = nNodes * (D + 1) + 2 * D;            // deg|agg|gsum|gsumsq
  k_zero<<<(zeroN + 255) / 256, 256, 0, stream>>>(deg, zeroN);

  k_wconv<<<((D * D / 4) + 255) / 256, 256, 0, stream>>>(W, wf);

  const long long scatterT = (long long)nEdges * 32;
  k_scatter<<<(int)((scatterT + 255) / 256), 256, 0, stream>>>(x, row, col, agg, deg, nEdges);

  const int n4 = nNodes * D / 4;
  k_h<<<(n4 + 255) / 256, 256, 0, stream>>>(x, agg, deg, hf, n4);

  k_gemm<<<nNodes / 16, 256, 0, stream>>>(hf, wf, b, y, gsum, gsumsq);

  k_bnparams<<<1, D, 0, stream>>>(gsum, gsumsq, gamma, beta, scale, shift, 1.0f / (float)nNodes);

  k_bnapply<<<(n4 + 255) / 256, 256, 0, stream>>>(y, scale, shift, n4);

  (void)n_in; (void)out_size; (void)ws_size;
}
